// Attention_Before_58463094833510
// MI455X (gfx1250) — matrix operand order repaired and checked by reading, not compiled, not run on hardware
//
#include <hip/hip_runtime.h>

typedef __attribute__((ext_vector_type(16))) __bf16        v16bf;
typedef __attribute__((ext_vector_type(8)))  float         v8f;
typedef __attribute__((ext_vector_type(8)))  unsigned int  v8u;
typedef __attribute__((ext_vector_type(4)))  int           v4i;
typedef __attribute__((ext_vector_type(8)))  int           v8i;
typedef __attribute__((ext_vector_type(4)))  unsigned int  v4u;

#define HIDDEN 1024
#define BM 128
#define BN 128
#define BK 32
#define KSTEPS (HIDDEN / BK)   // 32
#define LDAF 36                // fp32 elems per A row (32 + 4 pad) == TDM pad 4dw per 32dw
#define LDB  40                // bf16 elems per Bt row (32 + 8 pad -> 80B stride)

#if __has_builtin(__builtin_amdgcn_tensor_load_to_lds) && \
    __has_builtin(__builtin_amdgcn_s_wait_tensorcnt)
#define HAVE_TDM 1
#else
#define HAVE_TDM 0
#endif

#if __has_builtin(__builtin_amdgcn_global_load_async_to_lds_b128) && \
    __has_builtin(__builtin_amdgcn_s_wait_asynccnt)
#define HAVE_ASYNC 1
typedef __attribute__((address_space(1))) v4i GV4;
typedef __attribute__((address_space(3))) v4i LV4;
#else
#define HAVE_ASYNC 0
#endif

// Pack two fp32 -> two bf16 (round-half-up) in one dword: 2x v_add + 1x v_perm_b32.
__device__ __forceinline__ unsigned int pack2_bf16(float lo, float hi) {
    unsigned int ul = __builtin_bit_cast(unsigned int, lo) + 0x8000u;
    unsigned int uh = __builtin_bit_cast(unsigned int, hi) + 0x8000u;
    return __builtin_amdgcn_perm(uh, ul, 0x07060302u);
}

__device__ __forceinline__ unsigned short f2bf(float f) {
    return (unsigned short)((__builtin_bit_cast(unsigned int, f) + 0x8000u) >> 16);
}

#if HAVE_TDM
// Issue one TDM 2D tile load: X[m_base + 0..127][kk + 0..31] fp32 -> LDS rows of
// 32 dwords padded by 4 dwords (matches LDAF=36). Wave-uniform; call from one wave.
__device__ __forceinline__ void tdm_load_A(const float* __restrict__ X,
                                           unsigned lds_addr, int m_base, int kk) {
    unsigned long long ga =
        (unsigned long long)(size_t)(X + (size_t)m_base * HIDDEN + kk);
    v4u g0;
    g0[0] = 1u;                                            // count=1, user mode, no gather
    g0[1] = lds_addr;                                      // LDS byte address
    g0[2] = (unsigned)(ga & 0xffffffffu);                  // global_addr[31:0]
    g0[3] = (unsigned)((ga >> 32) & 0x01ffffffu)
          | 0x80000000u;                                   // global_addr[56:32] | type=2
    v8i g1;
    g1[0] = (int)(0x00020000u    // data_size = 2 (4 bytes)
          | (1u << 20)           // pad_enable
          | (4u << 22)           // pad_interval: 32 dwords
          | (3u << 25));         // pad_amount:   4 dwords
    g1[1] = (int)(1024u  << 16); // tensor_dim0 = 1024 (abar_addr=0)
    g1[2] = (int)(16384u << 16); // tensor_dim0 hi=0 | tensor_dim1 lo = 16384
    g1[3] = (int)(32u    << 16); // tensor_dim1 hi=0 | tile_dim0 = 32
    g1[4] = 128;                 // tile_dim1 = 128, tile_dim2 = 0
    g1[5] = 1024;                // tensor_dim0_stride = 1024
    g1[6] = 0;
    g1[7] = 0;
    v4i gz = (v4i)0;             // groups 2/3: unused for 2D tensor
#if defined(__clang_major__) && __clang_major__ >= 23
    __builtin_amdgcn_tensor_load_to_lds(g0, g1, gz, gz, (v8i)0, 0);
#else
    __builtin_amdgcn_tensor_load_to_lds(g0, g1, gz, gz, 0);
#endif
}
#else
// Stage A tile (128 x 32 fp32) raw into LDS: async DMA if available, else copy.
__device__ __forceinline__ void stage_A(const float* __restrict__ X, float* sAbuf,
                                        int m_base, int kk, int tid) {
    #pragma unroll
    for (int i = 0; i < 4; i++) {
        int g   = i * 256 + tid;
        int row = g >> 3;
        int q   = g & 7;
        const float* src = X + (size_t)(m_base + row) * HIDDEN + kk + q * 4;
        float*       dst = &sAbuf[row * LDAF + q * 4];
#if HAVE_ASYNC
        __builtin_amdgcn_global_load_async_to_lds_b128((GV4*)src, (LV4*)dst, 0, 0);
#else
        *(float4*)dst = *(const float4*)src;
#endif
    }
}
#endif

// Stage B tile transposed: W[k][n] fp32 -> sB[n][k] bf16.
__device__ __forceinline__ void stage_B(const float* __restrict__ Wp, unsigned short* sBbuf,
                                        int nc0, int kk, int tid) {
    #pragma unroll
    for (int i = 0; i < 4; i++) {
        int g  = i * 256 + tid;
        int kr = g >> 5;
        int q  = g & 31;
        float4 v = *(const float4*)(Wp + (size_t)(kk + kr) * HIDDEN + nc0 + q * 4);
        sBbuf[(q * 4 + 0) * LDB + kr] = f2bf(v.x);
        sBbuf[(q * 4 + 1) * LDB + kr] = f2bf(v.y);
        sBbuf[(q * 4 + 2) * LDB + kr] = f2bf(v.z);
        sBbuf[(q * 4 + 3) * LDB + kr] = f2bf(v.w);
    }
}

__global__ __launch_bounds__(256)
void qkv_wmma_kernel(const float* __restrict__ X,
                     const float* __restrict__ Wq, const float* __restrict__ bq,
                     const float* __restrict__ Wk, const float* __restrict__ bk,
                     const float* __restrict__ Wv, const float* __restrict__ bv,
                     float* __restrict__ out)
{
    __shared__ float          sA[2][BM * LDAF];   // raw fp32 A tiles (double buffered)
    __shared__ unsigned short sB[2][BN * LDB];    // bf16 transposed B tiles

    const int tid  = threadIdx.x;
    const int lane = tid & 31;
    const int wid  = tid >> 5;
    const int hf   = lane >> 4;
    const int l16  = lane & 15;

    const int bx = blockIdx.x;
    const int by = blockIdx.y;

    const int n_base = bx * BN;
    const int which  = n_base >> 10;
    const int nc0    = n_base & 1023;
    const float* __restrict__ Wp = (which == 0) ? Wq : ((which == 1) ? Wk : Wv);
    const float* __restrict__ bp = (which == 0) ? bq : ((which == 1) ? bk : bv);

    const int m_base = by * BM;
    const int wm = (wid & 3) * 32;
    const int wn = (wid >> 2) * 64;

#if HAVE_TDM
    const unsigned lds_a[2] = { (unsigned)(size_t)&sA[0][0],
                                (unsigned)(size_t)&sA[1][0] };
#endif

    v8f acc[2][4];
    #pragma unroll
    for (int i = 0; i < 2; i++)
        #pragma unroll
        for (int j = 0; j < 4; j++)
            acc[i][j] = (v8f)0.0f;

    // ---- pipeline prologue: fill buffer 0 ----
#if HAVE_TDM
    if (wid == 0) tdm_load_A(X, lds_a[0], m_base, 0);
#else
    stage_A(X, sA[0], m_base, 0, tid);
#endif
    stage_B(Wp, sB[0], nc0, 0, tid);
#if HAVE_TDM
    if (wid == 0) __builtin_amdgcn_s_wait_tensorcnt(0);
#elif HAVE_ASYNC
    __builtin_amdgcn_s_wait_asynccnt(0);
#endif
    __syncthreads();

    for (int it = 0; it < KSTEPS; ++it) {
        const int kk  = it * BK;
        const int cur = it & 1;

        // stage next tile into the idle buffer while computing on `cur`
        if (it + 1 < KSTEPS) {
#if HAVE_TDM
            if (wid == 0) tdm_load_A(X, lds_a[cur ^ 1], m_base, kk + BK);
#else
            stage_A(X, sA[cur ^ 1], m_base, kk + BK, tid);
#endif
            stage_B(Wp, sB[cur ^ 1], nc0, kk + BK, tid);
        }
        // prefetch two tiles ahead (global_prefetch_b8)
        if (kk + 2 * BK < HIDDEN) {
            if (tid < 128) {
                __builtin_prefetch(X + (size_t)(m_base + tid) * HIDDEN + kk + 2 * BK, 0, 1);
            } else {
                int t = tid - 128;
                __builtin_prefetch(Wp + (size_t)(kk + 2 * BK + (t >> 2)) * HIDDEN
                                      + nc0 + (t & 3) * 32, 0, 1);
            }
        }

        // ---- A fragments: fp32 LDS -> bf16 in-register (co-executes with WMMA) ----
        v16bf a[2], bfr[4];
        #pragma unroll
        for (int i = 0; i < 2; i++) {
            const float* arow = &sA[cur][(wm + i * 16 + l16) * LDAF];
            float4 f0 = *(const float4*)(arow + hf * 8);
            float4 f1 = *(const float4*)(arow + hf * 8 + 4);
            float4 f2 = *(const float4*)(arow + 16 + hf * 8);
            float4 f3 = *(const float4*)(arow + 16 + hf * 8 + 4);
            v8u p;
            p[0] = pack2_bf16(f0.x, f0.y); p[1] = pack2_bf16(f0.z, f0.w);
            p[2] = pack2_bf16(f1.x, f1.y); p[3] = pack2_bf16(f1.z, f1.w);
            p[4] = pack2_bf16(f2.x, f2.y); p[5] = pack2_bf16(f2.z, f2.w);
            p[6] = pack2_bf16(f3.x, f3.y); p[7] = pack2_bf16(f3.z, f3.w);
            a[i] = __builtin_bit_cast(v16bf, p);
        }
        // ---- B fragments: 16 bf16 per lane (K = hf*8 .. +7, 16 + hf*8 .. +7) ----
        #pragma unroll
        for (int j = 0; j < 4; j++) {
            const unsigned short* brow = &sB[cur][(wn + j * 16 + l16) * LDB + hf * 8];
            uint4 u0 = *(const uint4*)(brow);
            uint4 u1 = *(const uint4*)(brow + 16);
            v8u p;
            p[0] = u0.x; p[1] = u0.y; p[2] = u0.z; p[3] = u0.w;
            p[4] = u1.x; p[5] = u1.y; p[6] = u1.z; p[7] = u1.w;
            bfr[j] = __builtin_bit_cast(v16bf, p);
        }

        #pragma unroll
        for (int i = 0; i < 2; i++)
            #pragma unroll
            for (int j = 0; j < 4; j++)
                acc[i][j] = __builtin_amdgcn_wmma_f32_16x16x32_bf16(
                    false, a[i], false, bfr[j], (short)0, acc[i][j], false, false);

#if HAVE_TDM
        if (wid == 0) __builtin_amdgcn_s_wait_tensorcnt(0);
#elif HAVE_ASYNC
        __builtin_amdgcn_s_wait_asynccnt(0);
#endif
        __syncthreads();
    }

    // ---- epilogue: C/D layout m = r + 8*hf (VGPR r), n = lane&15 ----
    const int b_idx = m_base >> 12;
    #pragma unroll
    for (int i = 0; i < 2; i++) {
        #pragma unroll
        for (int j = 0; j < 4; j++) {
            int n_g = n_base + wn + j * 16 + l16;
            int h   = (n_g >> 6) & 15;
            int d   = n_g & 63;
            float bias = bp[n_g & 1023];
            #pragma unroll
            for (int r = 0; r < 8; r++) {
                int m_g = m_base + wm + i * 16 + r + 8 * hf;
                int s   = m_g & 4095;
                size_t ofs = (size_t)which * 16777216ull
                           + ((size_t)(b_idx * 16 + h) * 4096 + (size_t)s) * 64 + d;
                out[ofs] = acc[i][j][r] + bias;
            }
        }
    }
}

extern "C" void kernel_launch(void* const* d_in, const int* in_sizes, int n_in,
                              void* d_out, int out_size, void* d_ws, size_t ws_size,
                              hipStream_t stream) {
    const float* X  = (const float*)d_in[0];
    const float* Wq = (const float*)d_in[1];
    const float* bq = (const float*)d_in[2];
    const float* Wk = (const float*)d_in[3];
    const float* bk = (const float*)d_in[4];
    const float* Wv = (const float*)d_in[5];
    const float* bv = (const float*)d_in[6];
    float* out = (float*)d_out;

    dim3 grid(3072 / BN, 16384 / BM);   // (24, 128)
    qkv_wmma_kernel<<<grid, 256, 0, stream>>>(X, Wq, bq, Wk, bk, Wv, bv, out);
}
